// CrossNetGatingMixLayer_15960098472079
// MI455X (gfx1250) — matrix operand order repaired and checked by reading, compiled, not run on hardware
//
#include <hip/hip_runtime.h>
#include <hip/hip_bf16.h>
#include <math.h>

// Problem sizes (fixed by the reference)
#define B_N   16384
#define D_N   512
#define L_N   3
#define E_N   4
#define R_N   128

// Tiling
#define ROWS  64            // batch rows per workgroup (4 m-tiles of 16)
#define MT    4             // 16-row WMMA m-tiles per block
#define XP    520           // LDS pitch (bf16) for 512-wide tiles: 260 dwords == 4 (mod 64) -> conflict-free b128
#define HP    136           // LDS pitch (bf16) for 128-wide tiles: 68 dwords  == 4 (mod 64)
#define FRAGS_PER_LE 288    // 128 V-frags + 32 C-frags + 128 U-frags
#define FRAG_ELEMS   512    // 32 lanes x 16 bf16 per fragment (1 KB)

typedef __attribute__((ext_vector_type(16))) __bf16 v16bf;
typedef __attribute__((ext_vector_type(8)))  __bf16 v8bf;
typedef __attribute__((ext_vector_type(8)))  float  v8f;

static __device__ inline v8f vzero8() {
  v8f z;
#pragma unroll
  for (int j = 0; j < 8; ++j) z[j] = 0.0f;
  return z;
}

// CDNA5 hardware tanh (TRANS op). The ISA requires one independent op or
// V_NOP after a TRANS op before its destination is consumed; the compiler
// cannot see inside the asm to schedule that, so we include the v_nop.
static __device__ inline float fast_tanh(float x) {
  float r;
  asm("v_tanh_f32 %0, %1\n\tv_nop" : "=v"(r) : "v"(x));
  return r;
}

static __device__ inline v8f wmma_bf16(v16bf a, v16bf b, v8f c) {
  // D = A(16x32 bf16) x B(32x16 bf16) + C(16x16 f32)
  return __builtin_amdgcn_wmma_f32_16x16x32_bf16(
      /*neg_a=*/false, a, /*neg_b=*/false, b,
      /*c_mod=*/(short)0, c, /*reuse_a=*/false, /*reuse_b=*/false);
}

// A-fragment gather from an LDS tile (row-major, `pitch` bf16 per row).
// ISA 16-bit A layout: lanes 0-15 hold row M=lane, K = k0+{0..7, 16..23};
// lanes 16-31 hold row M=lane-16, K = k0+8+{0..7, 16..23}.
static __device__ inline v16bf load_a(const __bf16* base, int pitch,
                                      int mrow, int lrow, int khalf, int k0) {
  const __bf16* p = base + (mrow + lrow) * pitch + k0 + khalf;
  v16bf a;
  *(v8bf*)&a       = *(const v8bf*)(p);        // K = k0+khalf .. +7
  *((v8bf*)&a + 1) = *(const v8bf*)(p + 16);   // K = k0+khalf+16 .. +23
  return a;
}

// B-fragment load from packed global weights: lane-contiguous 32B per lane.
static __device__ inline v16bf load_b(const __bf16* frag, int lane) {
  return *(const v16bf*)(frag + lane * 16);
}

// ---------------------------------------------------------------------------
// Pack V/C/U fp32 weights into bf16 WMMA B-fragment order in d_ws.
// B layout (32x16 bf16): lane L holds column N=L%16, K = (L<16?0:8)+(i<8?i:i+8) for
// element i in 0..15 of the lane's 16-element chunk.
// Fragment order per (layer,expert): [ V: kt(16) x nt(8) | C: kt(4) x nt(8) |
//                                      U: kt(4) x nt(32) ], 512 bf16 each.
// ---------------------------------------------------------------------------
__global__ __launch_bounds__(256)
void pack_weights_kernel(const float* __restrict__ Us,
                         const float* __restrict__ Cs,
                         const float* __restrict__ Vs,
                         __bf16* __restrict__ wpack) {
  int gid = blockIdx.x * 256 + threadIdx.x;
  const int total = L_N * E_N * FRAGS_PER_LE * FRAG_ELEMS;
  if (gid >= total) return;

  int le   = gid / (FRAGS_PER_LE * FRAG_ELEMS);   // (layer*4 + expert)
  int rem  = gid % (FRAGS_PER_LE * FRAG_ELEMS);
  int frag = rem / FRAG_ELEMS;
  int ei   = rem % FRAG_ELEMS;
  int lane = ei >> 4;
  int i    = ei & 15;
  int kl   = ((lane >> 4) << 3) + (i & 7) + ((i >> 3) << 4);   // (lane<16 ? 0 : 8) + (i<8 ? i : 16+i-8)
  int nl   = lane & 15;

  float v;
  if (frag < 128) {
    // V: xv[b,r] = sum_d V[e,d,r] * x[b,d]  ->  B[k=d][n=r] = V[d][r]
    int kt = frag >> 3, nt = frag & 7;
    int d = kt * 32 + kl, r = nt * 16 + nl;
    v = Vs[((long)le * D_N + d) * R_N + r];
  } else if (frag < 160) {
    // C: t[b,r] = sum_s C[e,r,s] * h[b,s]   ->  B[k=s][n=r] = C[r][s]
    int f = frag - 128;
    int kt = f >> 3, nt = f & 7;
    int s = kt * 32 + kl, r = nt * 16 + nl;
    v = Cs[((long)le * R_N + r) * R_N + s];
  } else {
    // U: xu[b,d] = sum_r U[e,d,r] * t[b,r]  ->  B[k=r][n=d] = U[d][r]
    int f = frag - 160;
    int kt = f >> 5, nt = f & 31;
    int r = kt * 32 + kl, d = nt * 16 + nl;
    v = Us[((long)le * D_N + d) * R_N + r];
  }
  wpack[gid] = (__bf16)v;
}

// ---------------------------------------------------------------------------
// Fused 3-layer CrossNet-MoE. 128 threads = 4 waves, 64 batch rows per block.
// Wave w owns N-tiles {2w,2w+1} in the R=128 GEMMs and N-tiles {8w..8w+7} in
// the D=512 GEMM; every wave covers all 4 m-tiles (B-fragment reused 4x).
// ---------------------------------------------------------------------------
__global__ __launch_bounds__(128)
void crossnet_fused_kernel(const float* __restrict__ x,
                           const float* __restrict__ bias,
                           const __bf16* __restrict__ wpack,
                           float* __restrict__ out) {
  extern __shared__ __bf16 smem[];
  __bf16* xbf  = smem;                 // ROWS x XP   : xl tile (bf16)
  __bf16* htmp = xbf + ROWS * XP;      // ROWS x HP   : per-expert H staging
  __bf16* tst  = htmp + ROWS * HP;     // E_N x ROWS x HP : tanh(T) per expert

  const int tid   = threadIdx.x;
  const int lane  = tid & 31;
  const int w     = tid >> 5;
  const int lrow  = lane & 15;
  const int khalf = (lane >> 4) << 3;  // 0 for lanes 0-15, 8 for lanes 16-31
  const long rowbase = (long)blockIdx.x * ROWS;

  // Stage x tile into LDS as bf16 (coalesced global reads).
  for (int idx = tid; idx < ROWS * D_N; idx += 128) {
    int r = idx >> 9, c = idx & (D_N - 1);
    xbf[r * XP + c] = (__bf16)x[(rowbase + r) * D_N + c];
  }
  __syncthreads();

  for (int l = 0; l < L_N; ++l) {
    const __bf16* wl = wpack + (long)l * E_N * FRAGS_PER_LE * FRAG_ELEMS;

    for (int e = 0; e < E_N; ++e) {
      const __bf16* we = wl + (long)e * FRAGS_PER_LE * FRAG_ELEMS;

      // ---- Step 1: H = X(64x512) @ V[e](512x128); wave: 2 n-tiles x 4 m-tiles
      v8f acc[2][MT];
#pragma unroll
      for (int n = 0; n < 2; ++n)
#pragma unroll
        for (int m = 0; m < MT; ++m) acc[n][m] = vzero8();

      for (int kt = 0; kt < 16; ++kt) {
        v16bf a[MT];
#pragma unroll
        for (int m = 0; m < MT; ++m)
          a[m] = load_a(xbf, XP, m * 16, lrow, khalf, kt * 32);
#pragma unroll
        for (int n = 0; n < 2; ++n) {
          int nt = 2 * w + n;
          v16bf bf = load_b(we + (long)(kt * 8 + nt) * FRAG_ELEMS, lane);
#pragma unroll
          for (int m = 0; m < MT; ++m) acc[n][m] = wmma_bf16(a[m], bf, acc[n][m]);
        }
      }

      // tanh -> htmp (D-fragment layout: VGPR j -> row j + khalf)
#pragma unroll
      for (int n = 0; n < 2; ++n) {
        int col = (2 * w + n) * 16 + lrow;
#pragma unroll
        for (int m = 0; m < MT; ++m)
#pragma unroll
          for (int j = 0; j < 8; ++j)
            htmp[(m * 16 + j + khalf) * HP + col] = (__bf16)fast_tanh(acc[n][m][j]);
      }
      __syncthreads();

      // ---- Step 2: T = H(64x128) @ C[e]^T(128x128)
#pragma unroll
      for (int n = 0; n < 2; ++n)
#pragma unroll
        for (int m = 0; m < MT; ++m) acc[n][m] = vzero8();

      for (int kt = 0; kt < 4; ++kt) {
        v16bf a[MT];
#pragma unroll
        for (int m = 0; m < MT; ++m)
          a[m] = load_a(htmp, HP, m * 16, lrow, khalf, kt * 32);
#pragma unroll
        for (int n = 0; n < 2; ++n) {
          int nt = 2 * w + n;
          v16bf bf = load_b(we + (long)(128 + kt * 8 + nt) * FRAG_ELEMS, lane);
#pragma unroll
          for (int m = 0; m < MT; ++m) acc[n][m] = wmma_bf16(a[m], bf, acc[n][m]);
        }
      }
      __syncthreads();   // all htmp reads done before anyone rewrites htmp

      // tanh -> per-expert T store
      {
        __bf16* te = tst + (long)e * ROWS * HP;
#pragma unroll
        for (int n = 0; n < 2; ++n) {
          int col = (2 * w + n) * 16 + lrow;
#pragma unroll
          for (int m = 0; m < MT; ++m)
#pragma unroll
            for (int j = 0; j < 8; ++j)
              te[(m * 16 + j + khalf) * HP + col] = (__bf16)fast_tanh(acc[n][m][j]);
        }
      }
      __syncthreads();
    } // experts

    // ---- Step 3: S = sum_e T_e(64x128) @ U[e]^T(128x512), fused finalize.
    // N-tiles processed in pairs so each A-fragment LDS gather feeds 2 WMMAs
    // (~1 ds_load_b128 per WMMA, matching steps 1/2).
    const float* bl = bias + l * D_N;
    for (int ip = 0; ip < 4; ++ip) {
      v8f s[2][MT];
#pragma unroll
      for (int n2 = 0; n2 < 2; ++n2)
#pragma unroll
        for (int m = 0; m < MT; ++m) s[n2][m] = vzero8();

      for (int e = 0; e < E_N; ++e) {
        const __bf16* we = wl + (long)e * FRAGS_PER_LE * FRAG_ELEMS;
        const __bf16* te = tst + (long)e * ROWS * HP;
#pragma unroll
        for (int kt = 0; kt < 4; ++kt) {
          v16bf a[MT];
#pragma unroll
          for (int m = 0; m < MT; ++m)
            a[m] = load_a(te, HP, m * 16, lrow, khalf, kt * 32);
#pragma unroll
          for (int n2 = 0; n2 < 2; ++n2) {
            int ntg = w * 8 + ip * 2 + n2;   // global n-tile in 0..31
            v16bf bf = load_b(we + (long)(160 + kt * 32 + ntg) * FRAG_ELEMS, lane);
#pragma unroll
            for (int m = 0; m < MT; ++m) s[n2][m] = wmma_bf16(a[m], bf, s[n2][m]);
          }
        }
      }

      // Finalize: xl_new = x0 * (S + E*bias) + xl  (gates are all ones)
#pragma unroll
      for (int n2 = 0; n2 < 2; ++n2) {
        int ntg = w * 8 + ip * 2 + n2;
        int col = ntg * 16 + lrow;
        float bv = bl[col] * (float)E_N;
        if (l == L_N - 1) {
#pragma unroll
          for (int m = 0; m < MT; ++m)
#pragma unroll
            for (int j = 0; j < 8; ++j) {
              int row = m * 16 + j + khalf;
              float xl = (float)xbf[row * XP + col];
              float x0 = x[(rowbase + row) * D_N + col];
              out[(rowbase + row) * D_N + col] = x0 * (s[n2][m][j] + bv) + xl;
            }
        } else {
#pragma unroll
          for (int m = 0; m < MT; ++m)
#pragma unroll
            for (int j = 0; j < 8; ++j) {
              int row = m * 16 + j + khalf;
              float xl = (float)xbf[row * XP + col];
              float x0 = x[(rowbase + row) * D_N + col];
              xbf[row * XP + col] = (__bf16)(x0 * (s[n2][m][j] + bv) + xl);
            }
        }
      }
    }
    __syncthreads();   // next layer's step 1 reads the updated xbf
  } // layers
}

// ---------------------------------------------------------------------------
extern "C" void kernel_launch(void* const* d_in, const int* in_sizes, int n_in,
                              void* d_out, int out_size, void* d_ws, size_t ws_size,
                              hipStream_t stream) {
  const float* x  = (const float*)d_in[0];
  const float* Us = (const float*)d_in[1];
  const float* Cs = (const float*)d_in[2];
  const float* Vs = (const float*)d_in[3];
  // d_in[4] = G: unused (softmax over singleton axis -> gates are all ones)
  const float* b  = (const float*)d_in[5];
  float* out      = (float*)d_out;
  __bf16* wpack   = (__bf16*)d_ws;   // needs 3,538,944 bytes

  const int total = L_N * E_N * FRAGS_PER_LE * FRAG_ELEMS;
  pack_weights_kernel<<<(total + 255) / 256, 256, 0, stream>>>(Us, Cs, Vs, wpack);

  const size_t smem_bytes =
      (size_t)(ROWS * XP + ROWS * HP + E_N * ROWS * HP) * sizeof(__bf16); // 153,600 B
  crossnet_fused_kernel<<<B_N / ROWS, 128, smem_bytes, stream>>>(x, b, wpack, out);
}
